// Attention_59983513256141
// MI455X (gfx1250) — matrix operand order repaired and checked by reading, not compiled, not run on hardware
//
#include <hip/hip_runtime.h>
#include <hip/hip_bf16.h>

typedef _Float16 v16h __attribute__((ext_vector_type(16)));
typedef _Float16 v8h  __attribute__((ext_vector_type(8)));
typedef _Float16 v4h  __attribute__((ext_vector_type(4)));
typedef float    v8f  __attribute__((ext_vector_type(8)));

#define NHEAD 12
#define DHEAD 64
#define SEQN  1024
#define CDIM  768
#define BATCH 8

// ---------------------------------------------------------------------------
// Fragment loaders (CDNA5 wave32 WMMA 16x16x32 f16; contiguous k = 16*(lane>>4) + s in A and B).
// Callers pass a per-lane base pointer; constant offsets fold into the
// global_load 24-bit immediate.
// ---------------------------------------------------------------------------

// A-frag: p = base + (row0 + (lane&15))*ld + ((lane>>4)<<4); chunks at +0,+8.
static __device__ __forceinline__ v16h lda16(const _Float16* p) {
    v8h lo = *(const v8h*)(p);
    v8h hi = *(const v8h*)(p + 8);
    return __builtin_shufflevector(lo, hi, 0,1,2,3,4,5,6,7,8,9,10,11,12,13,14,15);
}

// B-frag from [N,K] row-major: p = base + (n0 + (lane&15))*ld + ((lane>>4)<<4).
static __device__ __forceinline__ v16h ldb16(const _Float16* p) {
    return *(const v16h*)(p);
}

static __device__ __forceinline__ v8f wmma32(v16h a, v16h b, v8f c) {
    return __builtin_amdgcn_wmma_f32_16x16x32_f16(false, a, false, b, (short)0, c, false, false);
}

// ---------------------------------------------------------------------------
// fp32 -> fp16 conversion, 4-wide
// ---------------------------------------------------------------------------
__global__ __launch_bounds__(256) void cvt_f32_f16_v4(const float4* __restrict__ src,
                                                      v4h* __restrict__ dst, int n4) {
    int i = blockIdx.x * blockDim.x + threadIdx.x;
    int stride = gridDim.x * blockDim.x;
    for (; i < n4; i += stride) {
        float4 s = src[i];
        v4h d;
        d[0] = (_Float16)s.x; d[1] = (_Float16)s.y;
        d[2] = (_Float16)s.z; d[3] = (_Float16)s.w;
        dst[i] = d;
    }
}

// ---------------------------------------------------------------------------
// QKV projection: [B*N, C] x [3C, C]^T -> Q[B,H,N,D], K[B,H,N,D], Vt[B,H,D,N].
// 16x64 tile per wave, double-buffered k-loop; unroll-2 ping-pong removes the
// fragment rotation copies. 512*36 = 18432 waves.
// ---------------------------------------------------------------------------
__global__ __launch_bounds__(256) void qkv_gemm(const _Float16* __restrict__ xh,
                                                const _Float16* __restrict__ wh,
                                                _Float16* __restrict__ qf,
                                                _Float16* __restrict__ kf,
                                                _Float16* __restrict__ vtf) {
    const int lane = threadIdx.x & 31;
    const int wave = blockIdx.x * (blockDim.x >> 5) + (threadIdx.x >> 5);
    const int mtile = wave / 36;
    const int cg    = wave - mtile * 36;
    const int m0 = mtile << 4;
    const int o0 = cg << 6;

    const _Float16* pa = xh + (size_t)(m0 + (lane & 15)) * CDIM + ((lane >> 4) << 4);
    const _Float16* pb = wh + (size_t)(o0 + (lane & 15)) * CDIM + ((lane >> 4) << 4);

    v8f acc[4] = {{}, {}, {}, {}};

    v16h a  = lda16(pa);
    v16h b0 = ldb16(pb);
    v16h b1 = ldb16(pb + 16 * CDIM);
    v16h b2 = ldb16(pb + 32 * CDIM);
    v16h b3 = ldb16(pb + 48 * CDIM);
    pa += 32; pb += 32;

#pragma unroll 2
    for (int it = 0; it < CDIM / 32 - 1; ++it) {
        v16h na  = lda16(pa);
        v16h nb0 = ldb16(pb);
        v16h nb1 = ldb16(pb + 16 * CDIM);
        v16h nb2 = ldb16(pb + 32 * CDIM);
        v16h nb3 = ldb16(pb + 48 * CDIM);
        pa += 32; pb += 32;

        acc[0] = wmma32(a, b0, acc[0]);
        acc[1] = wmma32(a, b1, acc[1]);
        acc[2] = wmma32(a, b2, acc[2]);
        acc[3] = wmma32(a, b3, acc[3]);

        a = na; b0 = nb0; b1 = nb1; b2 = nb2; b3 = nb3;
    }
    acc[0] = wmma32(a, b0, acc[0]);
    acc[1] = wmma32(a, b1, acc[1]);
    acc[2] = wmma32(a, b2, acc[2]);
    acc[3] = wmma32(a, b3, acc[3]);

    // scatter into Q / K / Vt (all control uniform per fragment)
    const int mbase = m0 + ((lane >> 4) << 3);   // 8 consecutive rows per lane
    const int bb    = mbase >> 10;               // constant across the 8 rows
    const int nn0   = mbase & 1023;
#pragma unroll
    for (int j = 0; j < 4; ++j) {
        const int o   = o0 + 16 * j + (lane & 15);
        const int sec = o / CDIM;
        const int oo  = o - sec * CDIM;
        const int h   = oo >> 6;
        const int d   = oo & 63;
        const size_t bh = (size_t)(bb * NHEAD + h);
        if (sec == 2) {
            // Vt rows are contiguous in nn -> one 16-byte store
            v8h pack;
#pragma unroll
            for (int v = 0; v < 8; ++v) pack[v] = (_Float16)acc[j][v];
            *(v8h*)&vtf[(bh * DHEAD + d) * SEQN + nn0] = pack;
        } else {
            _Float16* dst = (sec == 0) ? qf : kf;
#pragma unroll
            for (int v = 0; v < 8; ++v)
                dst[(bh * SEQN + (nn0 + v)) * DHEAD + d] = (_Float16)acc[j][v];
        }
    }
}

// ---------------------------------------------------------------------------
// Flash attention: one wave owns a 16-query tile of one (b,h); 32-key steps.
// K frags are refilled in place right after the S-WMMAs consume them; V frags
// issue before the softmax VALU/exp block. P->f16 via padded LDS (row stride
// 40 halves: bank-conflict-free). 6144 waves.
// ---------------------------------------------------------------------------
__global__ __launch_bounds__(256) void flash_attn(const _Float16* __restrict__ qf,
                                                  const _Float16* __restrict__ kf,
                                                  const _Float16* __restrict__ vtf,
                                                  _Float16* __restrict__ ao) {
    __shared__ __align__(16) _Float16 lds[8][16][40];
    const int lane = threadIdx.x & 31;
    const int w    = threadIdx.x >> 5;
    const int wave = blockIdx.x * 8 + w;
    const int bh   = wave >> 6;          // 64 q-tiles per (b,h)
    const int q0   = (wave & 63) << 4;

    const _Float16* qbase = qf  + (size_t)bh * SEQN * DHEAD;
    const _Float16* kbase = kf  + (size_t)bh * SEQN * DHEAD;
    const _Float16* vbase = vtf + (size_t)bh * DHEAD * SEQN;

    const _Float16* pq = qbase + (size_t)(q0 + (lane & 15)) * DHEAD + ((lane >> 4) << 4);
    const _Float16* pk = kbase + (size_t)(lane & 15) * DHEAD + ((lane >> 4) << 4);
    const _Float16* pv = vbase + (size_t)(lane & 15) * SEQN + ((lane >> 4) << 4);

    const v16h qa0 = lda16(pq);
    const v16h qa1 = lda16(pq + 32);

    v16h kf0 = ldb16(pk);
    v16h kf1 = ldb16(pk + 32);
    v16h kf2 = ldb16(pk + 16 * DHEAD);
    v16h kf3 = ldb16(pk + 16 * DHEAD + 32);
    pk += 32 * DHEAD;

    v8f oA = {}, oB = {}, oC = {}, oD = {};
    float mrow[8], lrow[8];
#pragma unroll
    for (int v = 0; v < 8; ++v) { mrow[v] = -1e30f; lrow[v] = 0.0f; }

    const float scale = 0.125f;  // 64^-0.5

#pragma unroll 1
    for (int it = 0; it < SEQN / 32; ++it) {
        // S = Q K^T (K frags resident from previous step)
        v8f s0 = {}, s1 = {};
        s0 = wmma32(qa0, kf0, s0);
        s0 = wmma32(qa1, kf1, s0);
        s1 = wmma32(qa0, kf2, s1);
        s1 = wmma32(qa1, kf3, s1);

        // refill K frags in place for the next step (last-iter overread lands
        // in the adjacent workspace region and is never consumed)
        kf0 = ldb16(pk);
        kf1 = ldb16(pk + 32);
        kf2 = ldb16(pk + 16 * DHEAD);
        kf3 = ldb16(pk + 16 * DHEAD + 32);
        pk += 32 * DHEAD;

        // V frags for this step, issued before the softmax math
        v16h vf0 = ldb16(pv);
        v16h vf1 = ldb16(pv + 16 * SEQN);
        v16h vf2 = ldb16(pv + 32 * SEQN);
        v16h vf3 = ldb16(pv + 48 * SEQN);
        pv += 32;

        // online softmax over the 32 new key columns (overlaps VMEM above)
#pragma unroll
        for (int v = 0; v < 8; ++v) {
            float t0 = s0[v] * scale;
            float t1 = s1[v] * scale;
            float mx = fmaxf(t0, t1);
            mx = fmaxf(mx, __shfl_xor(mx, 1));
            mx = fmaxf(mx, __shfl_xor(mx, 2));
            mx = fmaxf(mx, __shfl_xor(mx, 4));
            mx = fmaxf(mx, __shfl_xor(mx, 8));
            const float mnew  = fmaxf(mrow[v], mx);
            const float alpha = __expf(mrow[v] - mnew);
            const float p0 = __expf(t0 - mnew);
            const float p1 = __expf(t1 - mnew);
            float rs = p0 + p1;
            rs += __shfl_xor(rs, 1);
            rs += __shfl_xor(rs, 2);
            rs += __shfl_xor(rs, 4);
            rs += __shfl_xor(rs, 8);
            lrow[v] = lrow[v] * alpha + rs;
            mrow[v] = mnew;
            oA[v] *= alpha; oB[v] *= alpha; oC[v] *= alpha; oD[v] *= alpha;
            const int r = ((lane >> 4) << 3) + v;   // C/D layout row
            lds[w][r][lane & 15]        = (_Float16)p0;
            lds[w][r][16 + (lane & 15)] = (_Float16)p1;
        }

        // reload P (16x32) as an A-fragment; wave-private LDS region, same-wave
        // LDS ops are in-order -> no barrier needed.
        const int r  = lane & 15;
        const int c0 = (lane >> 4) << 4;
        v8h lo = *(const v8h*)&lds[w][r][c0];
        v8h hi = *(const v8h*)&lds[w][r][c0 + 8];
        v16h pa = __builtin_shufflevector(lo, hi, 0,1,2,3,4,5,6,7,8,9,10,11,12,13,14,15);

        oA = wmma32(pa, vf0, oA);
        oB = wmma32(pa, vf1, oB);
        oC = wmma32(pa, vf2, oC);
        oD = wmma32(pa, vf3, oD);
    }

    // epilogue: O / l -> ao[b, n, h*64 + d] (f16, feeds out-proj GEMM)
    const int bb = bh / NHEAD;
    const int h  = bh - NHEAD * bb;
    const int n  = lane & 15;
#pragma unroll
    for (int v = 0; v < 8; ++v) {
        const int m = q0 + ((lane >> 4) << 3) + v;
        const float inv = 1.0f / lrow[v];
        const size_t rowbase = ((size_t)bb * SEQN + m) * CDIM + h * DHEAD;
        ao[rowbase + 0  + n] = (_Float16)(oA[v] * inv);
        ao[rowbase + 16 + n] = (_Float16)(oB[v] * inv);
        ao[rowbase + 32 + n] = (_Float16)(oC[v] * inv);
        ao[rowbase + 48 + n] = (_Float16)(oD[v] * inv);
    }
}

// ---------------------------------------------------------------------------
// Output projection: [B*N, C] x [C, C]^T + bias -> fp32 d_out.
// 16x64 tile per wave, double-buffered, unroll-2 ping-pong. 512*12 waves.
// ---------------------------------------------------------------------------
__global__ __launch_bounds__(256) void proj_gemm(const _Float16* __restrict__ ah,
                                                 const _Float16* __restrict__ wh,
                                                 const float* __restrict__ bias,
                                                 float* __restrict__ out) {
    const int lane = threadIdx.x & 31;
    const int wave = blockIdx.x * (blockDim.x >> 5) + (threadIdx.x >> 5);
    const int mtile = wave / 12;
    const int cg    = wave - mtile * 12;
    const int m0 = mtile << 4;
    const int o0 = cg << 6;

    const _Float16* pa = ah + (size_t)(m0 + (lane & 15)) * CDIM + ((lane >> 4) << 4);
    const _Float16* pb = wh + (size_t)(o0 + (lane & 15)) * CDIM + ((lane >> 4) << 4);

    v8f acc[4] = {{}, {}, {}, {}};

    v16h a  = lda16(pa);
    v16h b0 = ldb16(pb);
    v16h b1 = ldb16(pb + 16 * CDIM);
    v16h b2 = ldb16(pb + 32 * CDIM);
    v16h b3 = ldb16(pb + 48 * CDIM);
    pa += 32; pb += 32;

#pragma unroll 2
    for (int it = 0; it < CDIM / 32 - 1; ++it) {
        v16h na  = lda16(pa);
        v16h nb0 = ldb16(pb);
        v16h nb1 = ldb16(pb + 16 * CDIM);
        v16h nb2 = ldb16(pb + 32 * CDIM);
        v16h nb3 = ldb16(pb + 48 * CDIM);
        pa += 32; pb += 32;

        acc[0] = wmma32(a, b0, acc[0]);
        acc[1] = wmma32(a, b1, acc[1]);
        acc[2] = wmma32(a, b2, acc[2]);
        acc[3] = wmma32(a, b3, acc[3]);

        a = na; b0 = nb0; b1 = nb1; b2 = nb2; b3 = nb3;
    }
    acc[0] = wmma32(a, b0, acc[0]);
    acc[1] = wmma32(a, b1, acc[1]);
    acc[2] = wmma32(a, b2, acc[2]);
    acc[3] = wmma32(a, b3, acc[3]);

    const int mbase = m0 + ((lane >> 4) << 3);
#pragma unroll
    for (int j = 0; j < 4; ++j) {
        const int o = o0 + 16 * j + (lane & 15);
        const float bo = bias[o];
#pragma unroll
        for (int v = 0; v < 8; ++v)
            out[(size_t)(mbase + v) * CDIM + o] = acc[j][v] + bo;
    }
}

// ---------------------------------------------------------------------------
// Host launcher
// ---------------------------------------------------------------------------
extern "C" void kernel_launch(void* const* d_in, const int* in_sizes, int n_in,
                              void* d_out, int out_size, void* d_ws, size_t ws_size,
                              hipStream_t stream) {
    (void)in_sizes; (void)n_in; (void)out_size; (void)ws_size;
    const float* x     = (const float*)d_in[0];   // [8,1024,768]
    const float* wqkv  = (const float*)d_in[1];   // [2304,768]
    const float* wproj = (const float*)d_in[2];   // [768,768]
    const float* bproj = (const float*)d_in[3];   // [768]
    float* out = (float*)d_out;                   // [8,1024,768]

    char* ws = (char*)d_ws;
    const size_t szXh    = (size_t)BATCH * SEQN * CDIM * 2;          // 12.6 MB
    const size_t szWqkv  = (size_t)3 * CDIM * CDIM * 2;              //  3.5 MB
    const size_t szWproj = (size_t)CDIM * CDIM * 2;                  //  1.2 MB
    const size_t szHead  = (size_t)BATCH * NHEAD * SEQN * DHEAD * 2; // 12.6 MB

    _Float16* xh     = (_Float16*)(ws);
    _Float16* wqkvh  = (_Float16*)(ws + szXh);
    _Float16* wprojh = (_Float16*)(ws + szXh + szWqkv);
    _Float16* qf     = (_Float16*)(ws + szXh + szWqkv + szWproj);
    _Float16* kf     = (_Float16*)(ws + szXh + szWqkv + szWproj + szHead);
    _Float16* vtf    = (_Float16*)(ws + szXh + szWqkv + szWproj + 2 * szHead);
    _Float16* ao     = (_Float16*)(ws + szXh + szWqkv + szWproj + 3 * szHead);

    cvt_f32_f16_v4<<<1536, 256, 0, stream>>>((const float4*)x,     (v4h*)xh,     BATCH * SEQN * CDIM / 4);
    cvt_f32_f16_v4<<<512,  256, 0, stream>>>((const float4*)wqkv,  (v4h*)wqkvh,  3 * CDIM * CDIM / 4);
    cvt_f32_f16_v4<<<256,  256, 0, stream>>>((const float4*)wproj, (v4h*)wprojh, CDIM * CDIM / 4);

    // 512 row-tiles * 36 col-groups = 18432 waves / 8 per block
    qkv_gemm<<<2304, 256, 0, stream>>>(xh, wqkvh, qf, kf, vtf);
    // 8*12*64 = 6144 q-tile waves / 8 per block
    flash_attn<<<768, 256, 0, stream>>>(qf, kf, vtf, ao);
    // 512 row-tiles * 12 col-groups = 6144 waves / 8 per block
    proj_gemm<<<768, 256, 0, stream>>>(ao, wprojh, bproj, out);
}
